// CRBM_6365141533015
// MI455X (gfx1250) — matrix operand order repaired and checked by reading, not compiled, not run on hardware
//
#include <hip/hip_runtime.h>
#include <hip/hip_bf16.h>

#define BDIM 64
#define DDIM 3072
#define HDIM 6144
#define KSTEPS 10
#define DCHUNK 128

typedef __attribute__((ext_vector_type(16))) __bf16 v16bf;
typedef __attribute__((ext_vector_type(8)))  float  v8f;
typedef unsigned int v4u __attribute__((ext_vector_type(4)));

union FragB { v16bf v; v4u q[2]; };

static __device__ __forceinline__ unsigned short f2bf_rne(float f) {
  unsigned u = __float_as_uint(f);
  unsigned r = u + 0x7FFFu + ((u >> 16) & 1u);
  return (unsigned short)(r >> 16);
}

// counter-based RNG: splitmix64 finalizer of (phase, element index)
static __device__ __forceinline__ float u01(unsigned phase, unsigned idx) {
  unsigned long long x = (((unsigned long long)phase) << 32) ^ (unsigned long long)idx
                         ^ 0x9E3779B97F4A7C15ULL;
  x ^= x >> 33; x *= 0xFF51AFD7ED558CCDULL;
  x ^= x >> 33; x *= 0xC4CEB9FE1A85EC53ULL;
  x ^= x >> 33;
  return (float)(unsigned)x * 2.3283064365386963e-10f; // [0,1)
}

// ---------------------------------------------------------------------------
// Pack W (fp32 DxH) into bf16 WMMA B-fragment layout for both orientations.
// Layout: [ntile][ktile][lane][16 bf16], lane = n%16 + 16*(k%32 >= 16),
// element index within lane = k%16  (contiguous k order, same as the A loads).
// ---------------------------------------------------------------------------
__global__ void pack_w_kernel(const float* __restrict__ W,
                              unsigned short* __restrict__ Wp,
                              unsigned short* __restrict__ WTp) {
  size_t idx = (size_t)blockIdx.x * blockDim.x + threadIdx.x;
  if (idx >= (size_t)DDIM * HDIM) return;
  unsigned d = (unsigned)(idx / HDIM);
  unsigned h = (unsigned)(idx % HDIM);
  unsigned short bf = f2bf_rne(W[idx]);
  { // GEMM1 (v @ W): K = d, N = h
    unsigned kt = d >> 5, kk = d & 31;
    unsigned nt = h >> 4, nn = h & 15;
    unsigned lane = nn + (kk & 16);
    size_t off = (((size_t)nt * (DDIM >> 5) + kt) * 32 + lane) * 16 + (kk & 15);
    Wp[off] = bf;
  }
  { // GEMM2 (h @ W^T): K = h, N = d
    unsigned kt = h >> 5, kk = h & 31;
    unsigned nt = d >> 4, nn = d & 15;
    unsigned lane = nn + (kk & 16);
    size_t off = (((size_t)nt * (HDIM >> 5) + kt) * 32 + lane) * 16 + (kk & 15);
    WTp[off] = bf;
  }
}

// a_hat = a + G*c ; v_bf = bf16(v0)
__global__ void prep_v_kernel(const int* __restrict__ v0, const float* __restrict__ c,
                              const float* __restrict__ a, const float* __restrict__ G,
                              float* __restrict__ a_hat, unsigned short* __restrict__ v_bf) {
  int idx = blockIdx.x * blockDim.x + threadIdx.x;
  if (idx >= BDIM * DDIM) return;
  int d = idx % DDIM;
  a_hat[idx] = a[d] + G[d] * c[idx];
  v_bf[idx] = (v0[idx] != 0) ? (unsigned short)0x3F80 : (unsigned short)0;
}

// b_hat = b + c @ F  (64 x H); F streamed once (coalesced along h), c staged in LDS.
__global__ __launch_bounds__(256)
void bhat_kernel(const float* __restrict__ c, const float* __restrict__ F,
                 const float* __restrict__ b, float* __restrict__ b_hat) {
  __shared__ float cs[BDIM * DCHUNK];
  int h = blockIdx.x * 256 + threadIdx.x;
  float acc[BDIM];
#pragma unroll
  for (int i = 0; i < BDIM; ++i) acc[i] = 0.f;
  for (int dc = 0; dc < DDIM; dc += DCHUNK) {
    for (int i = threadIdx.x; i < BDIM * DCHUNK; i += 256) {
      int bb = i / DCHUNK, dd = i % DCHUNK;
      cs[i] = c[(size_t)bb * DDIM + dc + dd];
    }
    __syncthreads();
    for (int dd = 0; dd < DCHUNK; ++dd) {
      float f = F[(size_t)(dc + dd) * HDIM + h];
#pragma unroll
      for (int bb = 0; bb < BDIM; ++bb)
        acc[bb] = fmaf(cs[bb * DCHUNK + dd], f, acc[bb]);
    }
    __syncthreads();
  }
  float bh = b[h];
  for (int bb = 0; bb < BDIM; ++bb)
    b_hat[(size_t)bb * HDIM + h] = acc[bb] + bh;
}

// ---------------------------------------------------------------------------
// One Gibbs half-step: out = bernoulli(sigmoid(clamp(bias + A @ Bpacked))).
// A: 64 x Kdim bf16 row-major. Bpacked: pre-striped B fragments.
// Each wave: one 16-row tile x two 16-col tiles (2 independent WMMA chains).
// Block = 8 waves covers 64 rows x 64 cols.
// ---------------------------------------------------------------------------
__global__ __launch_bounds__(256)
void gibbs_half_kernel(const unsigned short* __restrict__ A,
                       const unsigned short* __restrict__ Bp,
                       const float* __restrict__ bias,
                       unsigned short* __restrict__ out_bf,
                       float* __restrict__ out_f,
                       int Kdim, int N, unsigned phase, int write_f) {
  const int lane = threadIdx.x & 31;
  const int wave = threadIdx.x >> 5;
  const int mt   = wave & 3;        // row tile 0..3
  const int pair = wave >> 2;       // col-tile pair 0..1
  const int nt0  = blockIdx.x * 4 + pair * 2;

  const int row   = mt * 16 + (lane & 15);
  const int khalf = (lane & 16) ? 16 : 0;         // contiguous k: lane half h holds k = 16h..16h+15
  const unsigned short* ap = A + (size_t)row * Kdim + khalf;

  const int ktiles = Kdim >> 5;
  const unsigned short* bp0 = Bp + ((size_t)nt0 * ktiles * 32 + lane) * 16;
  const unsigned short* bp1 = bp0 + (size_t)ktiles * 512;

  v8f accs[2];
  accs[0] = (v8f){}; accs[1] = (v8f){};

  for (int kt = 0; kt < ktiles; ++kt) {
    FragB a, b0, b1;
    const unsigned short* ak  = ap  + kt * 32;
    const unsigned short* bk0 = bp0 + (size_t)kt * 512;
    const unsigned short* bk1 = bp1 + (size_t)kt * 512;
    a.q[0]  = *(const v4u*)(ak);
    a.q[1]  = *(const v4u*)(ak + 8);
    b0.q[0] = *(const v4u*)(bk0);
    b0.q[1] = *(const v4u*)(bk0 + 8);
    b1.q[0] = *(const v4u*)(bk1);
    b1.q[1] = *(const v4u*)(bk1 + 8);
    __builtin_prefetch(bk0 + 2048, 0, 1);   // global_prefetch_b8 on the W stream
    accs[0] = __builtin_amdgcn_wmma_f32_16x16x32_bf16(
        false, a.v, false, b0.v, (short)0, accs[0], false, false);
    accs[1] = __builtin_amdgcn_wmma_f32_16x16x32_bf16(
        false, a.v, false, b1.v, (short)0, accs[1], false, false);
  }

  // C layout: VGPR r -> M = r (lanes 0-15) / 8+r (lanes 16-31), N = lane%16
  const int mrow = mt * 16 + ((lane & 16) ? 8 : 0);
#pragma unroll
  for (int t = 0; t < 2; ++t) {
    const int col = (nt0 + t) * 16 + (lane & 15);
#pragma unroll
    for (int r = 0; r < 8; ++r) {
      int m = mrow + r;
      size_t o = (size_t)m * N + col;
      float z = accs[t][r] + bias[o];
      z = (z == z) ? z : 0.f;                       // nan_to_num
      z = fminf(30.f, fmaxf(-30.f, z));             // clamp
      float p = 1.f / (1.f + __expf(-z));           // sigmoid
      int s = (u01(phase, (unsigned)o) < p) ? 1 : 0;
      out_bf[o] = s ? (unsigned short)0x3F80 : (unsigned short)0;
      if (write_f) out_f[o] = (float)s;
    }
  }
}

extern "C" void kernel_launch(void* const* d_in, const int* in_sizes, int n_in,
                              void* d_out, int out_size, void* d_ws, size_t ws_size,
                              hipStream_t stream) {
  (void)in_sizes; (void)n_in; (void)out_size; (void)ws_size;
  const int*   v0 = (const int*)d_in[0];
  // d_in[1] = h0: dead in the reference (overwritten before first use)
  const float* c  = (const float*)d_in[2];
  const float* W  = (const float*)d_in[3];
  const float* a  = (const float*)d_in[4];
  const float* b  = (const float*)d_in[5];
  const float* F  = (const float*)d_in[6];
  const float* G  = (const float*)d_in[7];

  char* ws = (char*)d_ws;
  const size_t SZW = (size_t)DDIM * HDIM * 2;           // 37.7 MB each
  unsigned short* Wp   = (unsigned short*)ws; ws += SZW;
  unsigned short* WTp  = (unsigned short*)ws; ws += SZW;
  float*          bhat = (float*)ws;          ws += (size_t)BDIM * HDIM * 4;
  float*          ahat = (float*)ws;          ws += (size_t)BDIM * DDIM * 4;
  unsigned short* h_bf = (unsigned short*)ws; ws += (size_t)BDIM * HDIM * 2;
  unsigned short* v_bf = (unsigned short*)ws; ws += (size_t)BDIM * DDIM * 2;

  float* outV = (float*)d_out;                 // (B, D) then (B, H), return order
  float* outH = outV + (size_t)BDIM * DDIM;

  const size_t nW = (size_t)DDIM * HDIM;
  pack_w_kernel<<<(unsigned)((nW + 255) / 256), 256, 0, stream>>>(W, Wp, WTp);
  prep_v_kernel<<<(BDIM * DDIM + 255) / 256, 256, 0, stream>>>(v0, c, a, G, ahat, v_bf);
  bhat_kernel<<<HDIM / 256, 256, 0, stream>>>(c, F, b, bhat);

  for (int k = 0; k < KSTEPS; ++k) {
    const int last = (k == KSTEPS - 1);
    gibbs_half_kernel<<<HDIM / 64, 256, 0, stream>>>(
        v_bf, Wp, bhat, h_bf, outH, DDIM, HDIM, 2u * (unsigned)k + 1u, last);
    gibbs_half_kernel<<<DDIM / 64, 256, 0, stream>>>(
        h_bf, WTp, ahat, v_bf, outV, HDIM, DDIM, 2u * (unsigned)k + 2u, last);
  }
}
